// GCN_3238405341964
// MI455X (gfx1250) — matrix operand order repaired and checked by reading, not compiled, not run on hardware
//
#include <hip/hip_runtime.h>
#include <hip/hip_bf16.h>
#include <stdint.h>

// GCN (3x GCNConv + segment-max pool + BN + linear head) for MI455X / gfx1250.
//
// Design (reasoned from MI455X specs, compile-only):
//  - Dense transforms x@W use V_WMMA_F32_16X16X32_F16 (f16 in, f32 acc).
//  - A-tile (64x64 f32) staged into LDS by the Tensor Data Mover
//    (tensor_load_to_lds + s_wait_tensorcnt), overlapped with W staging.
//  - W staged into LDS in WMMA B-fragment order -> contiguous ds_load_b128
//    per lane instead of 128 strided ds_load_u16.
//  - Message passing dominates runtime: ~410MB gather + 102M f32 atomics per
//    layer; the ~110MB working set fits the 192MB L2, so edge-parallel
//    atomic scatter-add runs at L2 rates (wave32 spans 32 contiguous
//    features -> coalesced gather + same-cacheline atomics).
//  - segment_max via monotonic uint keys + atomicMax; BN + head in one block.

typedef __attribute__((ext_vector_type(16))) _Float16 v16h;
typedef __attribute__((ext_vector_type(8)))  float    v8f;
typedef __attribute__((ext_vector_type(4)))  unsigned int u32x4;
typedef __attribute__((ext_vector_type(8)))  int          i32x8;
typedef __attribute__((ext_vector_type(4)))  int          i32x4;

#define NFEAT 64

// ---------------------------------------------------------------- utilities
__global__ void fill_u32_kernel(unsigned* __restrict__ p, unsigned v, int n) {
  int i = blockIdx.x * blockDim.x + threadIdx.x;
  if (i < n) p[i] = v;
}

__device__ __forceinline__ unsigned f2key(float v) {
  unsigned b = __float_as_uint(v);
  return (b & 0x80000000u) ? ~b : (b | 0x80000000u);  // monotonic float->u32
}
__device__ __forceinline__ float key2f(unsigned k) {
  return __uint_as_float((k & 0x80000000u) ? (k ^ 0x80000000u) : ~k);
}

// ------------------------------------------------- WMMA GEMM: T = X @ W
// X: [nrows, 64] f32 row-major, W: [64, 64] f32 row-major, T: [nrows, 64] f32.
// Block = 128 threads (4 waves). Each wave computes a 16x64 output tile:
// 4 accumulators x (K-loop of 2 x 16x16x32 WMMA).
__global__ __launch_bounds__(128)
void gemm64_wmma_kernel(const float* __restrict__ X, const float* __restrict__ W,
                        float* __restrict__ T, int nrows) {
  // W in B-fragment order: sWf[kt][nt][lane*16 + j] = W_f16 for that lane's frag.
  __shared__ __align__(32) _Float16 sWf[2][4][32 * 16];  // 8 KB
  __shared__ __align__(16) float    sA[64 * NFEAT];      // 16 KB (block row tile)

  const int tid  = threadIdx.x;
  const int wave = tid >> 5;
  const int lane = tid & 31;
  const int row0 = blockIdx.x * 64;

  // ---- TDM: DMA the 64x64 f32 row tile of X into LDS (gfx1250 async tensor).
  // D# per CDNA5 ISA ch.8: 2D tile, data_size=4B, tile 64x64, dim0_stride=64.
  // Rows past (nrows - row0) are OOB -> TDM writes zeros (our padding).
  if (wave == 0) {
    unsigned long long ga = (unsigned long long)(uintptr_t)(X + (size_t)row0 * NFEAT);
    unsigned lds = (unsigned)(uintptr_t)(&sA[0]);        // LDS byte offset (addr[31:0])
    unsigned remRows = (unsigned)(nrows - row0);         // remaining tensor extent (dim1)
    u32x4 g0;
    g0[0] = 1u;                                          // count=1 (valid user D#)
    g0[1] = lds;                                         // lds_addr
    g0[2] = (unsigned)(ga & 0xFFFFFFFFu);                // global_addr[31:0]
    g0[3] = ((unsigned)(ga >> 32) & 0x01FFFFFFu) | (2u << 30);  // addr[56:32] | type=2
    i32x8 g1;
    g1[0] = (int)(2u << 16);                             // data_size = 4 bytes
    g1[1] = (int)(64u << 16);                            // tensor_dim0 = 64
    g1[2] = (int)((remRows & 0xFFFFu) << 16);            // tensor_dim1[15:0]
    g1[3] = (int)(((remRows >> 16) & 0xFFFFu) | (64u << 16)); // dim1[31:16] | tile_dim0=64
    g1[4] = 64;                                          // tile_dim1 = 64 rows
    g1[5] = 64;                                          // tensor_dim0_stride = 64
    g1[6] = 0;
    g1[7] = 0;
    i32x4 gz4 = {0, 0, 0, 0};                            // groups 2/3: unused (2D)
    i32x8 gz8 = {0, 0, 0, 0, 0, 0, 0, 0};
    __builtin_amdgcn_tensor_load_to_lds(g0, g1, gz4, gz4, gz8, 0);
  }

  // ---- Stage W (f32 -> f16) into fragment order while the TDM runs.
  // Lane ln of fragment (kt,nt) needs W[kt*32 + (ln>>4)*8 + (j&7) + (j>>3)*16][nt*16 + (ln&15)].
  for (int i = tid; i < 2 * 4 * 32 * 16; i += 128) {
    int kt = i >> 11, nt = (i >> 9) & 3, ln = (i >> 4) & 31, j = i & 15;
    int k = kt * 32 + (ln >> 4) * 8 + (j & 7) + ((j >> 3) << 4);
    int n = nt * 16 + (ln & 15);
    sWf[kt][nt][ln * 16 + j] = (_Float16)W[k * NFEAT + n];
  }

  if (wave == 0) __builtin_amdgcn_s_wait_tensorcnt(0);
  __syncthreads();

  const int mloc = wave * 16 + (lane & 15);  // row inside block tile
  const int hi   = lane >> 4;                // lane half-group
  v8f acc[4] = {};

  #pragma unroll
  for (int kt = 0; kt < 2; ++kt) {
    // A fragment (16x32 f16): lanes 0-15 hold K 0-7 & 16-23 (+kt*32),
    // lanes 16-31 hold K 8-15 & 24-31 (ISA 7.12.2 16-bit A layout).
    // Two contiguous 8-float runs -> ds_load_b128 pairs + v_cvt_pk_f16_f32.
    v16h a;
    const int kbase = kt * 32 + hi * 8;
    #pragma unroll
    for (int j = 0; j < 16; ++j) {
      int k = kbase + (j & 7) + ((j >> 3) << 4);
      a[j] = (_Float16)sA[mloc * NFEAT + k];
    }
    #pragma unroll
    for (int nt = 0; nt < 4; ++nt) {
      v16h b = *(const v16h*)&sWf[kt][nt][lane * 16];   // 32B contiguous
      acc[nt] = __builtin_amdgcn_wmma_f32_16x16x32_f16(
          false, a, false, b, (short)0, acc[nt], false, false);
    }
  }

  // C/D layout: VGPR v, lanes 0-15 -> M=v, lanes 16-31 -> M=v+8; N = lane&15.
  const int rbase = row0 + wave * 16 + hi * 8;
  const int col   = lane & 15;
  #pragma unroll
  for (int nt = 0; nt < 4; ++nt) {
    #pragma unroll
    for (int v = 0; v < 8; ++v) {
      int r = rbase + v;
      if (r < nrows) T[(size_t)r * NFEAT + nt * 16 + col] = acc[nt][v];
    }
  }
}

// ---------------------------------------- edge scatter: agg[dst] += ew*T[src]
// total = nEdges*64; thread = (edge e = idx>>6, feature f = idx&63).
__global__ void scatter_add_kernel(const float* __restrict__ T,
                                   const int* __restrict__ src,
                                   const int* __restrict__ dst,
                                   const float* __restrict__ ew,
                                   float* __restrict__ agg, int total) {
  int idx = blockIdx.x * blockDim.x + threadIdx.x;
  if (idx >= total) return;
  int e = idx >> 6, f = idx & 63;
  int s = src[e], d = dst[e];
  float v = ew[e] * T[(size_t)s * NFEAT + f];
  atomicAdd(agg + (size_t)d * NFEAT + f, v);
}

// ------------------------------------------------ h = (relu?)(h + b[f])
__global__ void bias_act_kernel(float* __restrict__ h, const float* __restrict__ b,
                                int total, int do_relu) {
  int idx = blockIdx.x * blockDim.x + threadIdx.x;
  if (idx >= total) return;
  float v = h[idx] + b[idx & 63];
  h[idx] = do_relu ? fmaxf(v, 0.0f) : v;
}

// -------------------------------------- segment max -> ordered-uint atomicMax
__global__ void pool_max_kernel(const float* __restrict__ h,
                                const int* __restrict__ batch,
                                unsigned* __restrict__ keys, int total) {
  int idx = blockIdx.x * blockDim.x + threadIdx.x;
  if (idx >= total) return;
  int node = idx >> 6, f = idx & 63;
  int g = batch[node];
  atomicMax(keys + (size_t)g * NFEAT + f, f2key(h[idx]));
}

// ------------------------------- BatchNorm (batch stats) + linear head [G,1]
__global__ __launch_bounds__(256)
void bn_head_kernel(const unsigned* __restrict__ keys,
                    const float* __restrict__ gamma, const float* __restrict__ beta,
                    const float* __restrict__ linW, const float* __restrict__ linb,
                    float* __restrict__ out, int ngraphs) {
  __shared__ float s_scale[NFEAT], s_shift[NFEAT];
  int t = threadIdx.x;
  if (t < NFEAT) {
    float m = 0.f, m2 = 0.f;
    for (int g = 0; g < ngraphs; ++g) {
      float v = key2f(keys[(size_t)g * NFEAT + t]);
      m += v; m2 += v * v;
    }
    float inv = 1.0f / (float)ngraphs;
    m *= inv;
    float var = fmaxf(m2 * inv - m * m, 0.0f);      // biased variance
    float sc  = rsqrtf(var + 1e-5f) * gamma[t];
    s_scale[t] = sc;
    s_shift[t] = beta[t] - m * sc;
  }
  __syncthreads();
  if (t < ngraphs) {
    float a = 0.f;
    #pragma unroll 4
    for (int f = 0; f < NFEAT; ++f) {
      float v = key2f(keys[(size_t)t * NFEAT + f]);
      a += (v * s_scale[f] + s_shift[f]) * linW[f];
    }
    out[t] = a + linb[0];
  }
}

// --------------------------------------------------------------------- driver
extern "C" void kernel_launch(void* const* d_in, const int* in_sizes, int n_in,
                              void* d_out, int out_size, void* d_ws, size_t ws_size,
                              hipStream_t stream) {
  const float* x     = (const float*)d_in[0];
  const int*   ei    = (const int*)d_in[1];   // [2, E]: row0 = src, row1 = dst
  const float* ew    = (const float*)d_in[2];
  const int*   batch = (const int*)d_in[3];
  const float* W1 = (const float*)d_in[4];
  const float* b1 = (const float*)d_in[5];
  const float* W2 = (const float*)d_in[6];
  const float* b2 = (const float*)d_in[7];
  const float* W3 = (const float*)d_in[8];
  const float* b3 = (const float*)d_in[9];
  const float* gamma = (const float*)d_in[10];
  const float* beta  = (const float*)d_in[11];
  const float* linW  = (const float*)d_in[12];
  const float* linb  = (const float*)d_in[13];
  float* out = (float*)d_out;

  const int N = in_sizes[0] / NFEAT;   // 100000 nodes
  const int E = in_sizes[2];           // 1600000 edges
  const int G = out_size;              // 256 graphs
  const int* src = ei;
  const int* dst = ei + E;

  // Workspace: T [N*64] f32 | H [N*64] f32 | pool keys [G*64] u32  (~51.3 MB)
  float*    bufT = (float*)d_ws;
  float*    bufH = bufT + (size_t)N * NFEAT;
  unsigned* keys = (unsigned*)(bufH + (size_t)N * NFEAT);

  const int nodeTot  = N * NFEAT;            // 6.4M
  const int edgeTot  = E * NFEAT;            // 102.4M (fits int32)
  const int gemmGrid = (N + 63) / 64;
  const int nodeGrid = (nodeTot + 255) / 256;
  const int edgeGrid = (edgeTot + 255) / 256;

  // ---- layer 1: h = relu(scatter(ew * (x@W1)[src]) + b1)
  gemm64_wmma_kernel<<<gemmGrid, 128, 0, stream>>>(x, W1, bufT, N);
  fill_u32_kernel<<<nodeGrid, 256, 0, stream>>>((unsigned*)bufH, 0u, nodeTot);
  scatter_add_kernel<<<edgeGrid, 256, 0, stream>>>(bufT, src, dst, ew, bufH, edgeTot);
  bias_act_kernel<<<nodeGrid, 256, 0, stream>>>(bufH, b1, nodeTot, 1);

  // ---- layer 2
  gemm64_wmma_kernel<<<gemmGrid, 128, 0, stream>>>(bufH, W2, bufT, N);
  fill_u32_kernel<<<nodeGrid, 256, 0, stream>>>((unsigned*)bufH, 0u, nodeTot);
  scatter_add_kernel<<<edgeGrid, 256, 0, stream>>>(bufT, src, dst, ew, bufH, edgeTot);
  bias_act_kernel<<<nodeGrid, 256, 0, stream>>>(bufH, b2, nodeTot, 1);

  // ---- layer 3 (no relu)
  gemm64_wmma_kernel<<<gemmGrid, 128, 0, stream>>>(bufH, W3, bufT, N);
  fill_u32_kernel<<<nodeGrid, 256, 0, stream>>>((unsigned*)bufH, 0u, nodeTot);
  scatter_add_kernel<<<edgeGrid, 256, 0, stream>>>(bufT, src, dst, ew, bufH, edgeTot);
  bias_act_kernel<<<nodeGrid, 256, 0, stream>>>(bufH, b3, nodeTot, 0);

  // ---- global max pool + BN + head
  fill_u32_kernel<<<(G * NFEAT + 255) / 256, 256, 0, stream>>>(keys, 0u, G * NFEAT);
  pool_max_kernel<<<nodeGrid, 256, 0, stream>>>(bufH, batch, keys, nodeTot);
  bn_head_kernel<<<1, 256, 0, stream>>>(keys, gamma, beta, linW, linb, out, G);
}
